// OTPartialOrderLoss_50362786512916
// MI455X (gfx1250) — matrix operand order repaired and checked by reading, not compiled, not run on hardware
//
#include <hip/hip_runtime.h>
#include <cstdint>

// ---------------- problem constants ----------------
#define BB   8192            // batch
#define NN   8213            // CAPTION_COUNT + B
#define CCAP 21
#define NP1  8224            // NN padded up to multiple of 32 (WMMA K-tiles)
#define M1C  0.05f
#define M2C  0.09f
#define PC   0.02f
#define NITER 50
// REG = -1/5  =>  K = exp(-M/REG) = exp(5*M), M = exp(-GM)

typedef __attribute__((ext_vector_type(16))) __bf16 v16bf;
typedef __attribute__((ext_vector_type(8)))  __bf16 v8bf;
typedef __attribute__((ext_vector_type(8)))  float  v8f;

__device__ __forceinline__ float relu(float x) { return fmaxf(x, 0.f); }

__device__ __forceinline__ float gm_of(float s, int lab) {
  if (lab == 0)      return relu(-PC - s) + relu(s);
  else if (lab == 3) return relu(M2C + s);
  else               return relu(M1C + s) + relu(-M2C - s);   // labels 1,2
}

// ---------------- setup kernels ----------------

// K1[i][j] = exp(5*exp(-GM_ij)) as bf16, zero-padded to NP1 columns.
__global__ void build_k1(const float* __restrict__ dist, const int* __restrict__ labels,
                         __bf16* __restrict__ K1) {
  int row = blockIdx.y;
  int j = blockIdx.x * blockDim.x + threadIdx.x;
  if (j >= NP1) return;
  size_t oidx = (size_t)row * NP1 + j;
  if (j >= NN) { K1[oidx] = (__bf16)0.f; return; }
  float d0 = dist[(size_t)row * NN];
  float s  = dist[(size_t)row * NN + j] - d0;
  int lab  = labels[(size_t)row * NN + j];
  float k  = __expf(5.f * __expf(-gm_of(s, lab)));
  K1[oidx] = (__bf16)k;
}

__global__ void build_diag(const float* __restrict__ dist, float* __restrict__ dg) {
  int i = blockIdx.x * blockDim.x + threadIdx.x;
  if (i < BB) dg[i] = dist[(size_t)i * NN + CCAP + i];
}

// K2: diag GM2 = 0 -> K = e^5 ; off-diag K = exp(5*exp(-relu(M2+s)))
__global__ void build_k2(const float* __restrict__ dist, const float* __restrict__ dg,
                         __bf16* __restrict__ K2) {
  int row = blockIdx.y;
  int j = blockIdx.x * blockDim.x + threadIdx.x;
  float k;
  if (j == row) {
    k = 148.4131591f;                      // exp(5)
  } else {
    float s = dist[(size_t)row * NN + CCAP + j] - dg[j];
    k = __expf(5.f * __expf(-relu(M2C + s)));
  }
  K2[(size_t)row * BB + j] = (__bf16)k;
}

__global__ void init_vecs(float* __restrict__ u1, float* __restrict__ v1,
                          float* __restrict__ u2, float* __restrict__ v2,
                          __bf16* __restrict__ vbf1) {
  int i = blockIdx.x * blockDim.x + threadIdx.x;
  if (i < BB) { u1[i] = 1.f / BB; u2[i] = 1.f / BB; v2[i] = 0.f; }
  if (i < NP1) { v1[i] = 0.f; vbf1[i] = (__bf16)0.f; }   // padded tail must stay 0
}

// ---------------- Sinkhorn iteration kernels ----------------

// v[j] = bval / sum_i K[i][j]*u[i]
// 4 columns per thread, one 8-byte load per row; bf16->f32 via pure bit shifts.
// Also writes a bf16 shadow of v for the WMMA row pass.
__global__ void col_update(const __bf16* __restrict__ K, const float* __restrict__ u,
                           float* __restrict__ v, __bf16* __restrict__ vbf,
                           int R, int C, int ld, float bval) {
  __shared__ float su[256];
  int j0 = (blockIdx.x * blockDim.x + threadIdx.x) * 4;
  bool active = (j0 < ld);                 // padded cols of K1 are zeros: loads safe
  int ldq = ld >> 2;                       // row stride in uint2 units
  float a0 = 0.f, a1 = 0.f, a2 = 0.f, a3 = 0.f;

  for (int i0 = 0; i0 < R; i0 += 256) {
    // stage u tile in LDS (async-to-LDS path when the gfx1250 builtins exist)
#if __has_builtin(__builtin_amdgcn_global_load_async_to_lds_b32) && \
    __has_builtin(__builtin_amdgcn_s_wait_asynccnt)
    __builtin_amdgcn_global_load_async_to_lds_b32(
        (__attribute__((address_space(1))) int*)(uintptr_t)(u + i0 + threadIdx.x),
        (__attribute__((address_space(3))) int*)(uintptr_t)(&su[threadIdx.x]),
        0, 0);
    __builtin_amdgcn_s_wait_asynccnt(0);
#else
    su[threadIdx.x] = u[i0 + threadIdx.x];
#endif
    __syncthreads();

    if (active) {
      const uint2* kp = (const uint2*)(K + (size_t)i0 * ld + j0);
      if (i0 + 256 < R)
        __builtin_prefetch(kp + (size_t)256 * ldq, 0, 0);   // global_prefetch_b8
      #pragma unroll 8
      for (int t = 0; t < 256; ++t) {
        uint2 w = kp[(size_t)t * ldq];
        float f0 = __uint_as_float(w.x << 16);
        float f1 = __uint_as_float(w.x & 0xffff0000u);
        float f2 = __uint_as_float(w.y << 16);
        float f3 = __uint_as_float(w.y & 0xffff0000u);
        float uu = su[t];
        a0 = fmaf(f0, uu, a0);
        a1 = fmaf(f1, uu, a1);
        a2 = fmaf(f2, uu, a2);
        a3 = fmaf(f3, uu, a3);
      }
    }
    __syncthreads();
  }

  if (j0     < C) { float r = bval / a0; v[j0]     = r; vbf[j0]     = (__bf16)r; }
  if (j0 + 1 < C) { float r = bval / a1; v[j0 + 1] = r; vbf[j0 + 1] = (__bf16)r; }
  if (j0 + 2 < C) { float r = bval / a2; v[j0 + 2] = r; vbf[j0 + 2] = (__bf16)r; }
  if (j0 + 3 < C) { float r = bval / a3; v[j0 + 3] = r; vbf[j0 + 3] = (__bf16)r; }
}

// u[rows] = aval / (K v) using v_wmma_f32_16x16x32_bf16.
// Each wave owns 16 rows; B operand = bf16 v segment replicated across the 16
// columns, so every D column holds the row dot; read it from the N=0 column.
__global__ void row_update_wmma(const __bf16* __restrict__ K, const __bf16* __restrict__ vbf,
                                float* __restrict__ u, int kdim, float aval) {
  int wave = threadIdx.x >> 5;
  int lane = threadIdx.x & 31;
  int half = lane >> 4;                 // ISA 16-bit A layout: lane half selects K group
  int mrow = lane & 15;
  int rowBase = blockIdx.x * 128 + wave * 16;

  const __bf16* rowp = K + (size_t)(rowBase + mrow) * kdim + half * 8;
  const __bf16* vp   = vbf + half * 8;
  v8f acc = {};

  for (int k0 = 0; k0 < kdim; k0 += 32) {
    // A fragment: lanes 0-15 hold K {0..7,16..23}, lanes 16-31 hold K {8..15,24..31}
    v8bf lo = *(const v8bf*)(rowp + k0);
    v8bf hi = *(const v8bf*)(rowp + k0 + 16);
    // B fragment: B[k][n] = v[k0+k] for all n (replicated columns), already bf16
    v8bf b0 = *(const v8bf*)(vp + k0);
    v8bf b1 = *(const v8bf*)(vp + k0 + 16);
    v16bf a, b;
    #pragma unroll
    for (int t = 0; t < 8; ++t) {
      a[t] = lo[t]; a[t + 8] = hi[t];
      b[t] = b0[t]; b[t + 8] = b1[t];
    }
    acc = __builtin_amdgcn_wmma_f32_16x16x32_bf16(
        /*neg_a=*/false, a, /*neg_b=*/false, b,
        /*c_mod=*/(short)0, acc, /*reuse_a=*/false, /*reuse_b=*/false);
  }

  // C/D layout: lane 0 (N=0) holds rows 0..7 in acc[0..7]; lane 16 holds rows 8..15.
  if ((lane & 15) == 0) {
    int roff = half * 8;
    #pragma unroll
    for (int m = 0; m < 8; ++m)
      u[rowBase + roff + m] = aval / acc[m];
  }
}

// ---------------- loss kernels ----------------
// T_norm[i][j] = K[i][j]*v[j] / S0_i ; u cancels. rowloss = S1/S0.

__global__ void loss1_rows(const float* __restrict__ dist, const int* __restrict__ labels,
                           const __bf16* __restrict__ K1, const float* __restrict__ v,
                           float* __restrict__ rowloss) {
  __shared__ float r0[256], r1[256];
  int i = blockIdx.x;
  float d0 = dist[(size_t)i * NN];
  float s0 = 0.f, s1 = 0.f;
  for (int j = threadIdx.x; j < NN; j += 256) {
    float kv = (float)K1[(size_t)i * NP1 + j] * v[j];
    float s  = dist[(size_t)i * NN + j] - d0;
    int lab  = labels[(size_t)i * NN + j];
    s0 += kv;
    s1 += gm_of(s, lab) * kv;
  }
  r0[threadIdx.x] = s0; r1[threadIdx.x] = s1;
  __syncthreads();
  for (int st = 128; st > 0; st >>= 1) {
    if (threadIdx.x < st) { r0[threadIdx.x] += r0[threadIdx.x + st];
                            r1[threadIdx.x] += r1[threadIdx.x + st]; }
    __syncthreads();
  }
  if (threadIdx.x == 0) rowloss[i] = r1[0] / r0[0];
}

__global__ void loss2_rows(const float* __restrict__ dist, const float* __restrict__ dg,
                           const __bf16* __restrict__ K2, const float* __restrict__ v,
                           float* __restrict__ rowloss) {
  __shared__ float r0[256], r1[256];
  int i = blockIdx.x;
  float s0 = 0.f, s1 = 0.f;
  for (int j = threadIdx.x; j < BB; j += 256) {
    if (j == i) continue;                  // diagonal of T2 zeroed before renorm
    float kv = (float)K2[(size_t)i * BB + j] * v[j];
    float s  = dist[(size_t)i * NN + CCAP + j] - dg[j];
    s0 += kv;
    s1 += relu(M2C + s) * kv;
  }
  r0[threadIdx.x] = s0; r1[threadIdx.x] = s1;
  __syncthreads();
  for (int st = 128; st > 0; st >>= 1) {
    if (threadIdx.x < st) { r0[threadIdx.x] += r0[threadIdx.x + st];
                            r1[threadIdx.x] += r1[threadIdx.x + st]; }
    __syncthreads();
  }
  if (threadIdx.x == 0) rowloss[i] = r1[0] / r0[0];
}

__global__ void final_reduce(const float* __restrict__ rl1, const float* __restrict__ rl2,
                             float* __restrict__ out) {
  __shared__ float sm[256];
  const float w1 = 1.f / ((float)BB * (float)NN);
  const float w2 = 1.f / ((float)BB * (float)BB);
  float acc = 0.f;
  for (int i = threadIdx.x; i < BB; i += 256)
    acc += rl1[i] * w1 + rl2[i] * w2;
  sm[threadIdx.x] = acc;
  __syncthreads();
  for (int st = 128; st > 0; st >>= 1) {
    if (threadIdx.x < st) sm[threadIdx.x] += sm[threadIdx.x + st];
    __syncthreads();
  }
  if (threadIdx.x == 0) out[0] = sm[0];
}

// ---------------- launcher ----------------
extern "C" void kernel_launch(void* const* d_in, const int* in_sizes, int n_in,
                              void* d_out, int out_size, void* d_ws, size_t ws_size,
                              hipStream_t stream) {
  const float* dist   = (const float*)d_in[0];
  const int*   labels = (const int*)d_in[1];
  float* out = (float*)d_out;

  // workspace carve-up (~269 MB)
  char* ws = (char*)d_ws;
  size_t off = 0;
  __bf16* K1   = (__bf16*)(ws + off); off += (size_t)BB * NP1 * sizeof(__bf16);
  __bf16* K2   = (__bf16*)(ws + off); off += (size_t)BB * BB  * sizeof(__bf16);
  float*  v1   = (float*)(ws + off);  off += (size_t)NP1 * sizeof(float);
  float*  u1   = (float*)(ws + off);  off += (size_t)BB  * sizeof(float);
  float*  v2   = (float*)(ws + off);  off += (size_t)BB  * sizeof(float);
  float*  u2   = (float*)(ws + off);  off += (size_t)BB  * sizeof(float);
  __bf16* vbf1 = (__bf16*)(ws + off); off += (size_t)NP1 * sizeof(__bf16);
  __bf16* vbf2 = (__bf16*)(ws + off); off += (size_t)BB  * sizeof(__bf16);
  float*  dg   = (float*)(ws + off);  off += (size_t)BB  * sizeof(float);
  float*  rl1  = (float*)(ws + off);  off += (size_t)BB  * sizeof(float);
  float*  rl2  = (float*)(ws + off);  off += (size_t)BB  * sizeof(float);

  dim3 thr(256);

  build_k1<<<dim3((NP1 + 255) / 256, BB), thr, 0, stream>>>(dist, labels, K1);
  build_diag<<<dim3(BB / 256), thr, 0, stream>>>(dist, dg);
  build_k2<<<dim3(BB / 256, BB), thr, 0, stream>>>(dist, dg, K2);
  init_vecs<<<dim3((NP1 + 255) / 256), thr, 0, stream>>>(u1, v1, u2, v2, vbf1);

  // Sinkhorn on K1 (B x N). Last u-update is unused by the loss (u cancels) -> skip.
  for (int it = 0; it < NITER; ++it) {
    col_update<<<dim3((NN + 1023) / 1024), thr, 0, stream>>>(
        K1, u1, v1, vbf1, BB, NN, NP1, 1.f / NN);
    if (it != NITER - 1)
      row_update_wmma<<<dim3(BB / 128), thr, 0, stream>>>(K1, vbf1, u1, NP1, 1.f / BB);
  }
  // Sinkhorn on K2 (B x B)
  for (int it = 0; it < NITER; ++it) {
    col_update<<<dim3(BB / 1024), thr, 0, stream>>>(
        K2, u2, v2, vbf2, BB, BB, BB, 1.f / BB);
    if (it != NITER - 1)
      row_update_wmma<<<dim3(BB / 128), thr, 0, stream>>>(K2, vbf2, u2, BB, 1.f / BB);
  }

  loss1_rows<<<dim3(BB), thr, 0, stream>>>(dist, labels, K1, v1, rl1);
  loss2_rows<<<dim3(BB), thr, 0, stream>>>(dist, dg, K2, v2, rl2);
  final_reduce<<<dim3(1), thr, 0, stream>>>(rl1, rl2, out);
}
